// NRAE_16423954940420
// MI455X (gfx1250) — matrix operand order repaired and checked by reading, not compiled, not run on hardware
//
#include <hip/hip_runtime.h>
#include <hip/hip_bf16.h>

typedef __attribute__((ext_vector_type(16))) __bf16 v16bf;
typedef __attribute__((ext_vector_type(8)))  __bf16 v8bf;
typedef __attribute__((ext_vector_type(8)))  float  v8f;

// GCC-style vector type matching the async-LDS builtin's expected pointee
typedef int v4i_vs __attribute__((vector_size(16)));
typedef __attribute__((address_space(1))) v4i_vs* gptr_b128;
typedef __attribute__((address_space(3))) v4i_vs* lptr_b128;

#define D_DIM 3072
#define H_DIM 1024
#define Z_DIM 32
#define BS_N  128
#define NN_N  32
#define NR_N  (BS_N * NN_N)   /* 4096 tangent rows */
#define MALL  (BS_N + NR_N)   /* 4224 encoder rows */
#define ZPAD  64              /* We3 padded column count */

#define BM 128
#define BN 64
#define BK 32
#define NTHREADS 256

#if __has_builtin(__builtin_amdgcn_global_load_async_to_lds_b128) && \
    __has_builtin(__builtin_amdgcn_s_wait_asynccnt)
#define HAVE_ASYNC_LDS 1
#else
#define HAVE_ASYNC_LDS 0
#endif

enum GemmMode { M_BIAS_RELU = 0, M_BIAS_F32 = 1, M_BIAS_RELU_MASKOUT = 2,
                M_MASKMUL = 3, M_LOSS = 4 };

struct GemmP {
  const __bf16* A;        // M x K, row-major bf16 (K % 32 == 0)
  const __bf16* B;        // K x N, row-major bf16 (N % 64 == 0, zero-padded)
  const float*  bias;     // Nout (fp32)
  const float*  maskIn;   // (M/rowDiv) x Nout fp32 0/1 mask
  int           rowDiv;   // output row -> mask/recon row divisor
  __bf16*       outBf;    // M x Nout bf16 output
  float*        outF;     // M x Nout f32 output
  float*        maskOut;  // M x Nout f32 mask output (pre-relu > 0)
  const float*  xTarget;  // M x Nout f32 (x_nn) for fused loss
  const float*  recon;    // (M/rowDiv) x Nout f32 for fused loss
  const float*  wRow;     // M f32 per-row weight (pre-scaled by 1/(BS*NN))
  float*        lossOut;  // scalar accumulator
  int M, N, K;            // N = padded compute width
  int Nout;               // true output width (store/bias pitch)
};

template <int MODE>
__global__ __launch_bounds__(NTHREADS)
void gemm_wmma_bf16(GemmP p) {
  __shared__ alignas(16) __bf16 As[BM][BK + 8];   // 128 x 40
  __shared__ alignas(16) __bf16 Bs[BN][BK + 8];   // 64 x 40 (transposed: [n][k])
  __shared__ float red[NTHREADS];

  const int tid  = threadIdx.x;
  const int wave = tid >> 5;
  const int lane = tid & 31;
  const int bm = blockIdx.x * BM;
  const int bn = blockIdx.y * BN;

  v8f acc[4] = {};

  // Fragment indexing per CDNA5 16-bit WMMA VGPR layouts (CDNA5 ISA)
  const int mA  = (wave << 4) + (lane & 15);  // A row within block tile
  const int kbA = (lane >> 4) << 3;           // 0 or 8 (A: K groups 0-7/8-15 then +16)
  const int kbB = (lane >> 4) << 3;           // 0 or 8 (B: K groups 0-7/8-15 then +16)
  const int nl  = lane & 15;

  const int kTiles = p.K / BK;
  for (int kt = 0; kt < kTiles; ++kt) {
    const int k0 = kt * BK;

    // ---- stage A tile (BM x BK): per-lane 16B global -> LDS copies ----
#if HAVE_ASYNC_LDS
#pragma unroll
    for (int c = 0; c < 2; ++c) {
      int chunk = tid + c * NTHREADS;          // 0..511
      int row = chunk >> 2;
      int col = (chunk & 3) << 3;
      __builtin_amdgcn_global_load_async_to_lds_b128(
          (gptr_b128)(p.A + (size_t)(bm + row) * p.K + k0 + col),
          (lptr_b128)&As[row][col],
          0, 0);
    }
#else
#pragma unroll
    for (int c = 0; c < 2; ++c) {
      int chunk = tid + c * NTHREADS;          // 0..511
      int row = chunk >> 2;
      int col = (chunk & 3) << 3;
      *reinterpret_cast<v8bf*>(&As[row][col]) =
          *reinterpret_cast<const v8bf*>(p.A + (size_t)(bm + row) * p.K + k0 + col);
    }
#endif

    // ---- stage B tile transposed: global [k][n] -> Bs[n][k] (no guard: padded) ----
    {
      int k  = tid >> 3;                       // 0..31
      int nv = (tid & 7) << 3;                 // 0..56
      v8bf bv = *reinterpret_cast<const v8bf*>(p.B + (size_t)(k0 + k) * p.N + bn + nv);
#pragma unroll
      for (int i = 0; i < 8; ++i) Bs[nv + i][k] = bv[i];
    }

    // prefetch next K tile while this one computes (emits global_prefetch_b8)
    if (kt + 1 < kTiles) {
      __builtin_prefetch(p.A + (size_t)(bm + (tid >> 1)) * p.K + k0 + BK + ((tid & 1) << 4), 0, 1);
      __builtin_prefetch(p.B + (size_t)(k0 + BK + (tid >> 3)) * p.N + bn + ((tid & 7) << 3), 0, 1);
    }

#if HAVE_ASYNC_LDS
    __builtin_amdgcn_s_wait_asynccnt(0);
#endif
    __syncthreads();

    // A fragment: lane holds M=mA, K pairs per layout
    v8bf alo = *reinterpret_cast<const v8bf*>(&As[mA][kbA]);
    v8bf ahi = *reinterpret_cast<const v8bf*>(&As[mA][kbA + 16]);
    v16bf afrag;
#pragma unroll
    for (int i = 0; i < 8; ++i) { afrag[i] = alo[i]; afrag[i + 8] = ahi[i]; }

#pragma unroll
    for (int j = 0; j < 4; ++j) {
      int n = (j << 4) + nl;
      v8bf blo = *reinterpret_cast<const v8bf*>(&Bs[n][kbB]);
      v8bf bhi = *reinterpret_cast<const v8bf*>(&Bs[n][kbB + 16]);
      v16bf bfrag;
#pragma unroll
      for (int i = 0; i < 8; ++i) { bfrag[i] = blo[i]; bfrag[i + 8] = bhi[i]; }
      acc[j] = __builtin_amdgcn_wmma_f32_16x16x32_bf16(
          false, afrag, false, bfrag, (short)0, acc[j], false, false);
    }
    __syncthreads();
  }

  // C/D layout: VGPR i -> M = i (lanes 0-15) / 8+i (lanes 16-31); N = lane&15
  const int r0 = bm + (wave << 4) + ((lane >> 4) << 3);

  if constexpr (MODE == M_LOSS) {
    float lsum = 0.0f;
#pragma unroll
    for (int j = 0; j < 4; ++j) {
      int n = bn + (j << 4) + nl;
#pragma unroll
      for (int i = 0; i < 8; ++i) {
        int m = r0 + i;
        float r = p.xTarget[(size_t)m * p.Nout + n]
                - p.recon[(size_t)(m / p.rowDiv) * p.Nout + n]
                - acc[j][i];
        lsum += p.wRow[m] * r * r;
      }
    }
    red[tid] = lsum;
    __syncthreads();
    for (int s = NTHREADS / 2; s > 0; s >>= 1) {
      if (tid < s) red[tid] += red[tid + s];
      __syncthreads();
    }
    if (tid == 0) atomicAdd(p.lossOut, red[0]);
  } else {
#pragma unroll
    for (int j = 0; j < 4; ++j) {
      int n = bn + (j << 4) + nl;
      if (n >= p.Nout) continue;               // only trims the padded N=32 layer
      float b = 0.0f;
      if constexpr (MODE != M_MASKMUL) b = p.bias[n];
#pragma unroll
      for (int i = 0; i < 8; ++i) {
        int m = r0 + i;
        float v = acc[j][i] + b;
        if constexpr (MODE == M_BIAS_RELU) {
          p.outBf[(size_t)m * p.Nout + n] = (__bf16)fmaxf(v, 0.0f);
        } else if constexpr (MODE == M_BIAS_F32) {
          p.outF[(size_t)m * p.Nout + n] = v;
        } else if constexpr (MODE == M_BIAS_RELU_MASKOUT) {
          p.maskOut[(size_t)m * p.Nout + n] = (v > 0.0f) ? 1.0f : 0.0f;
          p.outBf[(size_t)m * p.Nout + n] = (__bf16)fmaxf(v, 0.0f);
        } else { // M_MASKMUL: dh = relu-mask(center) * (v @ W)
          float mk = p.maskIn[(size_t)(m / p.rowDiv) * p.Nout + n];
          p.outBf[(size_t)m * p.Nout + n] = (__bf16)(acc[j][i] * mk);
        }
      }
    }
  }
}

__global__ void cast_f32_bf16(const float* __restrict__ src,
                              __bf16* __restrict__ dst, int n) {
  int i = blockIdx.x * blockDim.x + threadIdx.x;
  if (i < n) dst[i] = (__bf16)src[i];
}

// cast rows x cols fp32 into rows x pitch bf16, zero-padding cols..pitch-1
__global__ void cast_pad_bf16(const float* __restrict__ src,
                              __bf16* __restrict__ dst,
                              int rows, int cols, int pitch) {
  int i = blockIdx.x * blockDim.x + threadIdx.x;
  if (i >= rows * pitch) return;
  int r = i / pitch, c = i - r * pitch;
  dst[i] = (c < cols) ? (__bf16)src[(size_t)r * cols + c] : (__bf16)0.0f;
}

// zcb = bf16(z_c); V = bf16(z_nn - z_c)
__global__ void prep_vz(const float* __restrict__ Zall,
                        __bf16* __restrict__ zcb, __bf16* __restrict__ V) {
  int i = blockIdx.x * blockDim.x + threadIdx.x;
  if (i < BS_N * Z_DIM) { zcb[i] = (__bf16)Zall[i]; return; }
  int j = i - BS_N * Z_DIM;
  if (j >= NR_N * Z_DIM) return;
  int row = j >> 5;          // tangent row 0..4095
  int z   = j & 31;
  int c   = row >> 5;        // center 0..127
  V[j] = (__bf16)(Zall[(size_t)(BS_N + row) * Z_DIM + z]
                - Zall[(size_t)c * Z_DIM + z]);
}

// binary kernel weight, pre-scaled by 1/(BS*NN) so atomic accumulates the mean
__global__ void weights_kernel(const float* __restrict__ xc,
                               const float* __restrict__ xnn,
                               float* __restrict__ w) {
  __shared__ float red[NTHREADS];
  int i = blockIdx.x;                     // 0..4095
  int c = i >> 5;
  float s = 0.0f;
  for (int d = threadIdx.x; d < D_DIM; d += NTHREADS) {
    float diff = xc[(size_t)c * D_DIM + d] - xnn[(size_t)i * D_DIM + d];
    s += diff * diff;
  }
  red[threadIdx.x] = s;
  __syncthreads();
  for (int off = NTHREADS / 2; off > 0; off >>= 1) {
    if (threadIdx.x < off) red[threadIdx.x] += red[threadIdx.x + off];
    __syncthreads();
  }
  if (threadIdx.x == 0) {
    float dist = sqrtf(red[0]);
    w[i] = (dist > 1e-12f ? 1.0f : 0.5f) * (1.0f / (float)(NR_N));
  }
}

__global__ void zero_scalar(float* p) { p[0] = 0.0f; }

extern "C" void kernel_launch(void* const* d_in, const int* in_sizes, int n_in,
                              void* d_out, int out_size, void* d_ws, size_t ws_size,
                              hipStream_t stream) {
  const float* x_c  = (const float*)d_in[0];
  const float* x_nn = (const float*)d_in[1];
  const float* We1  = (const float*)d_in[2];
  const float* be1  = (const float*)d_in[3];
  const float* We2  = (const float*)d_in[4];
  const float* be2  = (const float*)d_in[5];
  const float* We3  = (const float*)d_in[6];
  const float* be3  = (const float*)d_in[7];
  const float* Wd1  = (const float*)d_in[8];
  const float* bd1  = (const float*)d_in[9];
  const float* Wd2  = (const float*)d_in[10];
  const float* bd2  = (const float*)d_in[11];
  const float* Wd3  = (const float*)d_in[12];
  const float* bd3  = (const float*)d_in[13];
  float* out = (float*)d_out;

  char* ws = (char*)d_ws;
  size_t off = 0;
  auto alloc = [&](size_t bytes) -> void* {
    void* p = ws + off;
    off = (off + bytes + 255) & ~(size_t)255;
    return p;
  };

  __bf16* Xall  = (__bf16*)alloc((size_t)MALL * D_DIM * 2);
  __bf16* We1b  = (__bf16*)alloc((size_t)D_DIM * H_DIM * 2);
  __bf16* We2b  = (__bf16*)alloc((size_t)H_DIM * H_DIM * 2);
  __bf16* We3b  = (__bf16*)alloc((size_t)H_DIM * ZPAD * 2);   // zero-padded to 64 cols
  __bf16* Wd1b  = (__bf16*)alloc((size_t)Z_DIM * H_DIM * 2);
  __bf16* Wd2b  = (__bf16*)alloc((size_t)H_DIM * H_DIM * 2);
  __bf16* Wd3b  = (__bf16*)alloc((size_t)H_DIM * D_DIM * 2);
  __bf16* H1    = (__bf16*)alloc((size_t)MALL * H_DIM * 2);
  __bf16* H2    = (__bf16*)alloc((size_t)MALL * H_DIM * 2);
  float*  Zall  = (float*)alloc((size_t)MALL * Z_DIM * 4);
  __bf16* zcb   = (__bf16*)alloc((size_t)BS_N * Z_DIM * 2);
  __bf16* Vb    = (__bf16*)alloc((size_t)NR_N * Z_DIM * 2);
  __bf16* h1d   = (__bf16*)alloc((size_t)BS_N * H_DIM * 2);
  __bf16* h2d   = (__bf16*)alloc((size_t)BS_N * H_DIM * 2);
  float*  mask1 = (float*)alloc((size_t)BS_N * H_DIM * 4);
  float*  mask2 = (float*)alloc((size_t)BS_N * H_DIM * 4);
  float*  recon = (float*)alloc((size_t)BS_N * D_DIM * 4);
  __bf16* dh1   = (__bf16*)alloc((size_t)NR_N * H_DIM * 2);
  __bf16* dh2   = (__bf16*)alloc((size_t)NR_N * H_DIM * 2);
  float*  wrow  = (float*)alloc((size_t)NR_N * 4);

  zero_scalar<<<1, 1, 0, stream>>>(out);

  auto cast = [&](const float* s, __bf16* d, int n) {
    cast_f32_bf16<<<(n + NTHREADS - 1) / NTHREADS, NTHREADS, 0, stream>>>(s, d, n);
  };
  cast(x_c,  Xall, BS_N * D_DIM);
  cast(x_nn, Xall + (size_t)BS_N * D_DIM, NR_N * D_DIM);
  cast(We1, We1b, D_DIM * H_DIM);
  cast(We2, We2b, H_DIM * H_DIM);
  cast_pad_bf16<<<(H_DIM * ZPAD + NTHREADS - 1) / NTHREADS, NTHREADS, 0, stream>>>(
      We3, We3b, H_DIM, Z_DIM, ZPAD);
  cast(Wd1, Wd1b, Z_DIM * H_DIM);
  cast(Wd2, Wd2b, H_DIM * H_DIM);
  cast(Wd3, Wd3b, H_DIM * D_DIM);

  weights_kernel<<<NR_N, NTHREADS, 0, stream>>>(x_c, x_nn, wrow);

  GemmP p;
  p.maskIn = nullptr; p.rowDiv = 1; p.outBf = nullptr; p.outF = nullptr;
  p.maskOut = nullptr; p.xTarget = nullptr; p.recon = nullptr;
  p.wRow = nullptr; p.lossOut = nullptr; p.bias = nullptr;

  // Encoder L1: Xall @ We1 + be1, relu -> H1
  p.A = Xall; p.B = We1b; p.bias = be1; p.outBf = H1;
  p.M = MALL; p.N = H_DIM; p.Nout = H_DIM; p.K = D_DIM;
  gemm_wmma_bf16<M_BIAS_RELU><<<dim3(MALL / BM, H_DIM / BN), NTHREADS, 0, stream>>>(p);

  // Encoder L2: H1 @ We2 + be2, relu -> H2
  p.A = H1; p.B = We2b; p.bias = be2; p.outBf = H2;
  p.M = MALL; p.N = H_DIM; p.Nout = H_DIM; p.K = H_DIM;
  gemm_wmma_bf16<M_BIAS_RELU><<<dim3(MALL / BM, H_DIM / BN), NTHREADS, 0, stream>>>(p);

  // Encoder L3: H2 @ We3(padded) + be3 -> Zall (f32, pitch 32)
  p.A = H2; p.B = We3b; p.bias = be3; p.outBf = nullptr; p.outF = Zall;
  p.M = MALL; p.N = ZPAD; p.Nout = Z_DIM; p.K = H_DIM;
  gemm_wmma_bf16<M_BIAS_F32><<<dim3(MALL / BM, 1), NTHREADS, 0, stream>>>(p);

  // z_c -> bf16; V = z_nn - z_c -> bf16
  {
    int total = BS_N * Z_DIM + NR_N * Z_DIM;
    prep_vz<<<(total + NTHREADS - 1) / NTHREADS, NTHREADS, 0, stream>>>(Zall, zcb, Vb);
  }

  // Decoder primal L1: zcb @ Wd1 + bd1, relu -> h1d ; mask1
  p.A = zcb; p.B = Wd1b; p.bias = bd1; p.outBf = h1d; p.outF = nullptr;
  p.maskOut = mask1; p.M = BS_N; p.N = H_DIM; p.Nout = H_DIM; p.K = Z_DIM;
  gemm_wmma_bf16<M_BIAS_RELU_MASKOUT><<<dim3(1, H_DIM / BN), NTHREADS, 0, stream>>>(p);

  // Decoder primal L2: h1d @ Wd2 + bd2, relu -> h2d ; mask2
  p.A = h1d; p.B = Wd2b; p.bias = bd2; p.outBf = h2d; p.maskOut = mask2;
  p.M = BS_N; p.N = H_DIM; p.Nout = H_DIM; p.K = H_DIM;
  gemm_wmma_bf16<M_BIAS_RELU_MASKOUT><<<dim3(1, H_DIM / BN), NTHREADS, 0, stream>>>(p);

  // Decoder primal L3: h2d @ Wd3 + bd3 -> recon (f32)
  p.A = h2d; p.B = Wd3b; p.bias = bd3; p.outBf = nullptr; p.maskOut = nullptr;
  p.outF = recon; p.M = BS_N; p.N = D_DIM; p.Nout = D_DIM; p.K = H_DIM;
  gemm_wmma_bf16<M_BIAS_F32><<<dim3(1, D_DIM / BN), NTHREADS, 0, stream>>>(p);

  // Tangent L1: dh1 = mask1[center] * (V @ Wd1)
  p.A = Vb; p.B = Wd1b; p.bias = nullptr; p.outF = nullptr; p.outBf = dh1;
  p.maskIn = mask1; p.rowDiv = NN_N; p.M = NR_N; p.N = H_DIM; p.Nout = H_DIM; p.K = Z_DIM;
  gemm_wmma_bf16<M_MASKMUL><<<dim3(NR_N / BM, H_DIM / BN), NTHREADS, 0, stream>>>(p);

  // Tangent L2: dh2 = mask2[center] * (dh1 @ Wd2)
  p.A = dh1; p.B = Wd2b; p.outBf = dh2; p.maskIn = mask2;
  p.M = NR_N; p.N = H_DIM; p.Nout = H_DIM; p.K = H_DIM;
  gemm_wmma_bf16<M_MASKMUL><<<dim3(NR_N / BM, H_DIM / BN), NTHREADS, 0, stream>>>(p);

  // Tangent L3 + fused loss: jac = dh2 @ Wd3 ;
  // loss += w[row] * (x_nn - recon[center] - jac)^2  (hess term is identically 0)
  p.A = dh2; p.B = Wd3b; p.outBf = nullptr; p.maskIn = nullptr;
  p.xTarget = x_nn; p.recon = recon; p.wRow = wrow; p.lossOut = out;
  p.rowDiv = NN_N; p.M = NR_N; p.N = D_DIM; p.Nout = D_DIM; p.K = H_DIM;
  gemm_wmma_bf16<M_LOSS><<<dim3(NR_N / BM, D_DIM / BN), NTHREADS, 0, stream>>>(p);
}
